// MultiHeadAttention_89721866814008
// MI455X (gfx1250) — matrix operand order repaired and checked by reading, not compiled, not run on hardware
//
#include <hip/hip_runtime.h>
#include <math.h>

// MultiHeadAttention forward for MI455X (gfx1250, wave32, WMMA).
// B=2, S=2048, D=768, H=12, DK=64.
// d_out = [output (B,S,D) f32 | attn_weights (B,H,S,S) f32]
// d_ws  = Qh | Kh | Vt | ctx | Xq | Xk | Xv | Wq | Wk | Wv | Wo  (all f16, ~47 MiB)

#define S_LEN 2048
#define D_DIM 768
#define NH    12
#define DK    64
#define BATCH 2

typedef _Float16 h8   __attribute__((ext_vector_type(8)));
typedef _Float16 v16h __attribute__((ext_vector_type(16)));
typedef float    v8f  __attribute__((ext_vector_type(8)));
typedef float    f8   __attribute__((ext_vector_type(8)));
typedef float    f4   __attribute__((ext_vector_type(4)));

union V16H { v16h v; h8 h[2]; };

__device__ __forceinline__ v8f wmma_f16(v16h a, v16h b, v8f c) {
  return __builtin_amdgcn_wmma_f32_16x16x32_f16(false, a, false, b,
                                                (short)0, c, false, false);
}

// reductions across a 16-lane half-wave (row groups of the C layout)
__device__ __forceinline__ float rmax16(float x) {
  x = fmaxf(x, __shfl_xor(x, 1, 32));
  x = fmaxf(x, __shfl_xor(x, 2, 32));
  x = fmaxf(x, __shfl_xor(x, 4, 32));
  x = fmaxf(x, __shfl_xor(x, 8, 32));
  return x;
}
__device__ __forceinline__ float rsum16(float x) {
  x += __shfl_xor(x, 1, 32);
  x += __shfl_xor(x, 2, 32);
  x += __shfl_xor(x, 4, 32);
  x += __shfl_xor(x, 8, 32);
  return x;
}

__device__ __forceinline__ h8 cvt_h8(f8 f) {
  h8 r;
#pragma unroll
  for (int i = 0; i < 8; i++) r[i] = (_Float16)f[i];
  return r;
}

// ---------------------------------------------------------------------------
// One-pass f32 -> f16 conversion (8 elements/thread). n must be multiple of
// 2048 per launch config below.
// ---------------------------------------------------------------------------
__global__ __launch_bounds__(256) void cvt_kernel(const float* __restrict__ src,
                                                  _Float16* __restrict__ dst) {
  const size_t i = ((size_t)blockIdx.x * 256 + threadIdx.x) * 8;
  *(h8*)(dst + i) = cvt_h8(*(const f8*)(src + i));
}

// ---------------------------------------------------------------------------
// Q/K projection: Y = X @ W^T  (M=4096, N=768, K=768), all-f16 operands,
// output per-head f16 (B,H,S,DK). Wave tile 32x64, 8 waves -> 128x128/block.
// ---------------------------------------------------------------------------
__global__ __launch_bounds__(256) void qk_proj_kernel(
    const _Float16* __restrict__ xq, const _Float16* __restrict__ xk,
    const _Float16* __restrict__ wq, const _Float16* __restrict__ wk,
    _Float16* __restrict__ q_out, _Float16* __restrict__ k_out) {
  const _Float16* X = (blockIdx.z == 0) ? xq : xk;
  const _Float16* W = (blockIdx.z == 0) ? wq : wk;
  _Float16* Y       = (blockIdx.z == 0) ? q_out : k_out;

  const int lane = threadIdx.x & 31;
  const int wib  = threadIdx.x >> 5;
  const int half = lane >> 4, ln = lane & 15;
  const int m_base = blockIdx.x * 128 + (wib & 3) * 32;
  const int n_base = blockIdx.y * 128 + (wib >> 2) * 64;

  v8f acc[2][4] = {};

  for (int kb = 0; kb < D_DIM; kb += 32) {
    V16H a[2];
#pragma unroll
    for (int r = 0; r < 2; r++) {
      const _Float16* xr = X + (size_t)(m_base + r * 16 + ln) * D_DIM + kb + half * 16;
      a[r].h[0] = *(const h8*)(xr);
      a[r].h[1] = *(const h8*)(xr + 8);
    }
#pragma unroll
    for (int f = 0; f < 4; f++) {
      const v16h bv = *(const v16h*)(W + (size_t)(n_base + f * 16 + ln) * D_DIM +
                                     kb + half * 16);
#pragma unroll
      for (int r = 0; r < 2; r++) acc[r][f] = wmma_f16(a[r].v, bv, acc[r][f]);
    }
  }

#pragma unroll
  for (int r = 0; r < 2; r++)
#pragma unroll
    for (int f = 0; f < 4; f++) {
      const int n = n_base + f * 16 + ln;
      const int hh = n >> 6, dk = n & 63;
#pragma unroll
      for (int v = 0; v < 8; v++) {
        const int mrow = m_base + r * 16 + half * 8 + v;
        const int bb = mrow >> 11, ss = mrow & 2047;
        Y[(((size_t)(bb * NH + hh)) * S_LEN + ss) * DK + dk] = (_Float16)acc[r][f][v];
      }
    }
}

// ---------------------------------------------------------------------------
// V projection, transposed: Vt[n,s] = sum_k w_v[n,k] * v_in[s,k]
// M-dim = channel n (768), N-dim = sequence s (4096). Output (B,H,DK,S) f16.
// ---------------------------------------------------------------------------
__global__ __launch_bounds__(256) void v_proj_kernel(
    const _Float16* __restrict__ xv, const _Float16* __restrict__ wv,
    _Float16* __restrict__ vt) {
  const int lane = threadIdx.x & 31;
  const int wib  = threadIdx.x >> 5;
  const int half = lane >> 4, ln = lane & 15;
  const int m_base = blockIdx.x * 128 + (wib & 3) * 32;   // channel dim
  const int n_base = blockIdx.y * 128 + (wib >> 2) * 64;  // sequence dim

  v8f acc[2][4] = {};

  for (int kb = 0; kb < D_DIM; kb += 32) {
    V16H a[2];
#pragma unroll
    for (int r = 0; r < 2; r++) {
      const _Float16* wr = wv + (size_t)(m_base + r * 16 + ln) * D_DIM + kb + half * 16;
      a[r].h[0] = *(const h8*)(wr);
      a[r].h[1] = *(const h8*)(wr + 8);
    }
#pragma unroll
    for (int f = 0; f < 4; f++) {
      const v16h bv = *(const v16h*)(xv + (size_t)(n_base + f * 16 + ln) * D_DIM +
                                     kb + half * 16);
#pragma unroll
      for (int r = 0; r < 2; r++) acc[r][f] = wmma_f16(a[r].v, bv, acc[r][f]);
    }
  }

#pragma unroll
  for (int r = 0; r < 2; r++)
#pragma unroll
    for (int f = 0; f < 4; f++) {
      const int s = n_base + f * 16 + ln;
      const int bb = s >> 11, sl = s & 2047;
#pragma unroll
      for (int v = 0; v < 8; v++) {
        const int ch = m_base + r * 16 + half * 8 + v;
        const int hh = ch >> 6, dk = ch & 63;
        vt[(((size_t)(bb * NH + hh)) * DK + dk) * S_LEN + sl] = (_Float16)acc[r][f][v];
      }
    }
}

// ---------------------------------------------------------------------------
// Fused causal attention. Each wave owns 16 query rows of one (b,h):
//   pass 1: flash (online max/sum) + context accumulation via WMMA,
//           P staged through LDS to convert C-layout -> A-layout.
//   pass 2: recompute scores and write normalized weights; vectorized
//           float4 zero-fill for the strictly-masked tail.
// ---------------------------------------------------------------------------
__global__ __launch_bounds__(128) void attn_kernel(
    const _Float16* __restrict__ Qh, const _Float16* __restrict__ Kh,
    const _Float16* __restrict__ Vt, float* __restrict__ attn,
    _Float16* __restrict__ ctx) {
  __shared__ _Float16 pstage[4][16 * 40];  // padded stride 40 halves

  const int lane = threadIdx.x & 31;
  const int wib  = threadIdx.x >> 5;
  const int wid  = blockIdx.x * 4 + wib;
  const int qt = wid & 127;            // S/16 = 128
  const int hh = (wid >> 7) % NH;
  const int bb = wid / (128 * NH);
  const int q0 = qt * 16;
  const int half = lane >> 4, ln = lane & 15;
  _Float16* ps = pstage[wib];

  const _Float16* qb  = Qh + (size_t)(bb * NH + hh) * S_LEN * DK;
  const _Float16* kb_ = Kh + (size_t)(bb * NH + hh) * S_LEN * DK;
  const _Float16* vb  = Vt + (size_t)(bb * NH + hh) * DK * S_LEN;

  // Q A-fragments for dk 0..31 and 32..63, resident for the whole kernel
  V16H aq0, aq1;
  {
    const _Float16* qr = qb + (size_t)(q0 + ln) * DK + half * 16;
    aq0.h[0] = *(const h8*)(qr);
    aq0.h[1] = *(const h8*)(qr + 8);
    aq1.h[0] = *(const h8*)(qr + 32);
    aq1.h[1] = *(const h8*)(qr + 40);
  }

  float m[8], lsum[8];
#pragma unroll
  for (int v = 0; v < 8; v++) { m[v] = -__builtin_inff(); lsum[v] = 0.f; }
  v8f o0 = {}, o1 = {}, o2 = {}, o3 = {};
  const float scale = 0.125f;  // 1/sqrt(DK)
  // all chunks kb in [0, kend) contain at least one unmasked column
  const int kend = ((q0 + 16 + 31) >> 5) << 5;

  // ---- pass 1: flash accumulation over causal keys ----
  for (int kb = 0; kb < kend; kb += 32) {
    v8f c0 = {}, c1 = {};
    {
      const _Float16* kr = kb_ + (size_t)(kb + ln) * DK + half * 16;
      c0 = wmma_f16(aq0.v, *(const v16h*)(kr), c0);
      c0 = wmma_f16(aq1.v, *(const v16h*)(kr + 32), c0);
      const _Float16* kr2 = kb_ + (size_t)(kb + 16 + ln) * DK + half * 16;
      c1 = wmma_f16(aq0.v, *(const v16h*)(kr2), c1);
      c1 = wmma_f16(aq1.v, *(const v16h*)(kr2 + 32), c1);
    }
    const int key0 = kb + ln, key1 = kb + 16 + ln;
    float e0[8], e1[8], fr[8];
#pragma unroll
    for (int v = 0; v < 8; v++) {
      const int qrow = q0 + half * 8 + v;
      const float s0 = (key0 <= qrow) ? c0[v] * scale : -__builtin_inff();
      const float s1 = (key1 <= qrow) ? c1[v] * scale : -__builtin_inff();
      const float tm = rmax16(fmaxf(s0, s1));
      const float nm = fmaxf(m[v], tm);
      const float f  = __expf(m[v] - nm);  // 0 when m was -inf
      e0[v] = __expf(s0 - nm);
      e1[v] = __expf(s1 - nm);
      lsum[v] = lsum[v] * f + rsum16(e0[v] + e1[v]);
      m[v] = nm;
      fr[v] = f;
    }
#pragma unroll
    for (int v = 0; v < 8; v++) {
      o0[v] *= fr[v]; o1[v] *= fr[v]; o2[v] *= fr[v]; o3[v] *= fr[v];
    }
    // stage P (C layout -> row-major LDS), then read back in A layout
#pragma unroll
    for (int v = 0; v < 8; v++) {
      const int row = half * 8 + v;
      ps[row * 40 + ln]      = (_Float16)e0[v];
      ps[row * 40 + 16 + ln] = (_Float16)e1[v];
    }
    asm volatile("s_wait_dscnt 0" ::: "memory");
    V16H ap;
    ap.h[0] = *(const h8*)(ps + ln * 40 + half * 16);
    ap.h[1] = *(const h8*)(ps + ln * 40 + half * 16 + 8);
    const _Float16* vr = vb + (size_t)ln * S_LEN + kb + half * 16;
    o0 = wmma_f16(ap.v, *(const v16h*)(vr), o0);
    o1 = wmma_f16(ap.v, *(const v16h*)(vr + 16 * S_LEN), o1);
    o2 = wmma_f16(ap.v, *(const v16h*)(vr + 32 * S_LEN), o2);
    o3 = wmma_f16(ap.v, *(const v16h*)(vr + 48 * S_LEN), o3);
  }

  // normalize + store context (layout (B,S,H*DK) for the output projection)
  float rl[8];
#pragma unroll
  for (int v = 0; v < 8; v++) rl[v] = 1.0f / lsum[v];
  {
    _Float16* cb = ctx + ((size_t)bb * S_LEN + q0) * D_DIM + hh * DK;
#pragma unroll
    for (int v = 0; v < 8; v++) {
      const int row = half * 8 + v;
      _Float16* cr = cb + (size_t)row * D_DIM + ln;
      cr[0]  = (_Float16)(o0[v] * rl[v]);
      cr[16] = (_Float16)(o1[v] * rl[v]);
      cr[32] = (_Float16)(o2[v] * rl[v]);
      cr[48] = (_Float16)(o3[v] * rl[v]);
    }
  }

  // ---- pass 2: write normalized attention weights ----
  float* ab = attn + (size_t)(bb * NH + hh) * S_LEN * S_LEN + (size_t)q0 * S_LEN;
  for (int kb = 0; kb < kend; kb += 32) {
    v8f c0 = {}, c1 = {};
    const _Float16* kr = kb_ + (size_t)(kb + ln) * DK + half * 16;
    c0 = wmma_f16(aq0.v, *(const v16h*)(kr), c0);
    c0 = wmma_f16(aq1.v, *(const v16h*)(kr + 32), c0);
    const _Float16* kr2 = kb_ + (size_t)(kb + 16 + ln) * DK + half * 16;
    c1 = wmma_f16(aq0.v, *(const v16h*)(kr2), c1);
    c1 = wmma_f16(aq1.v, *(const v16h*)(kr2 + 32), c1);
#pragma unroll
    for (int v = 0; v < 8; v++) {
      const int row = half * 8 + v;
      const int qrow = q0 + row;
      float p0 = 0.f, p1 = 0.f;
      if (kb + ln <= qrow)      p0 = __expf(c0[v] * scale - m[v]) * rl[v];
      if (kb + 16 + ln <= qrow) p1 = __expf(c1[v] * scale - m[v]) * rl[v];
      ab[(size_t)row * S_LEN + kb + ln]      = p0;
      ab[(size_t)row * S_LEN + kb + 16 + ln] = p1;
    }
  }
  // strictly-masked tail: vectorized zero-fill (cols [kend, S))
  {
    const f4 z = {};
#pragma unroll 1
    for (int r = 0; r < 16; r++) {
      float* rowp = ab + (size_t)r * S_LEN;
      for (int c = kend + lane * 4; c < S_LEN; c += 128) {
        *(f4*)(rowp + c) = z;
      }
    }
  }
}

// ---------------------------------------------------------------------------
// Output projection: out = ctx(f16) @ w_o^T, f32 result straight to d_out.
// ---------------------------------------------------------------------------
__global__ __launch_bounds__(256) void o_proj_kernel(
    const _Float16* __restrict__ ctx, const _Float16* __restrict__ wo,
    float* __restrict__ out) {
  const int lane = threadIdx.x & 31;
  const int wib  = threadIdx.x >> 5;
  const int half = lane >> 4, ln = lane & 15;
  const int m_base = blockIdx.x * 128 + (wib & 3) * 32;
  const int n_base = blockIdx.y * 128 + (wib >> 2) * 64;

  v8f acc[2][4] = {};

  for (int kb = 0; kb < D_DIM; kb += 32) {
    V16H a[2];
#pragma unroll
    for (int r = 0; r < 2; r++) {
      const _Float16* cr = ctx + (size_t)(m_base + r * 16 + ln) * D_DIM + kb + half * 16;
      a[r].h[0] = *(const h8*)(cr);
      a[r].h[1] = *(const h8*)(cr + 8);
    }
#pragma unroll
    for (int f = 0; f < 4; f++) {
      const v16h bv = *(const v16h*)(wo + (size_t)(n_base + f * 16 + ln) * D_DIM +
                                     kb + half * 16);
#pragma unroll
      for (int r = 0; r < 2; r++) acc[r][f] = wmma_f16(a[r].v, bv, acc[r][f]);
    }
  }

#pragma unroll
  for (int r = 0; r < 2; r++)
#pragma unroll
    for (int f = 0; f < 4; f++) {
      const int n = n_base + f * 16 + ln;
#pragma unroll
      for (int v = 0; v < 8; v++) {
        const int mrow = m_base + r * 16 + half * 8 + v;
        out[(size_t)mrow * D_DIM + n] = acc[r][f][v];
      }
    }
}

// ---------------------------------------------------------------------------
extern "C" void kernel_launch(void* const* d_in, const int* in_sizes, int n_in,
                              void* d_out, int out_size, void* d_ws, size_t ws_size,
                              hipStream_t stream) {
  (void)in_sizes; (void)n_in; (void)out_size; (void)ws_size;
  const float* q_in = (const float*)d_in[0];
  const float* k_in = (const float*)d_in[1];
  const float* v_in = (const float*)d_in[2];
  // d_in[3] is the causal triu mask; reproduced analytically in attn_kernel.
  const float* w_q = (const float*)d_in[4];
  const float* w_k = (const float*)d_in[5];
  const float* w_v = (const float*)d_in[6];
  const float* w_o = (const float*)d_in[7];

  float* out  = (float*)d_out;                                  // (B,S,D)
  float* attn = out + (size_t)BATCH * S_LEN * D_DIM;            // (B,H,S,S)

  const size_t perBHS = (size_t)BATCH * NH * S_LEN * DK;        // 3,145,728
  const size_t perX   = (size_t)BATCH * S_LEN * D_DIM;          // 3,145,728
  const size_t perW   = (size_t)D_DIM * D_DIM;                  //   589,824

  _Float16* Qh  = (_Float16*)d_ws;
  _Float16* Kh  = Qh + perBHS;
  _Float16* Vt  = Kh + perBHS;
  _Float16* Ctx = Vt + perBHS;
  _Float16* Xq  = Ctx + perX;
  _Float16* Xk  = Xq + perX;
  _Float16* Xv  = Xk + perX;
  _Float16* Wq  = Xv + perX;
  _Float16* Wk  = Wq + perW;
  _Float16* Wv  = Wk + perW;
  _Float16* Wo  = Wv + perW;   // needs ws_size >= ~47 MiB

  const int xblk = (int)(perX / 2048);  // 1536
  const int wblk = (int)(perW / 2048);  //  288
  cvt_kernel<<<xblk, 256, 0, stream>>>(q_in, Xq);
  cvt_kernel<<<xblk, 256, 0, stream>>>(k_in, Xk);
  cvt_kernel<<<xblk, 256, 0, stream>>>(v_in, Xv);
  cvt_kernel<<<wblk, 256, 0, stream>>>(w_q, Wq);
  cvt_kernel<<<wblk, 256, 0, stream>>>(w_k, Wk);
  cvt_kernel<<<wblk, 256, 0, stream>>>(w_v, Wv);
  cvt_kernel<<<wblk, 256, 0, stream>>>(w_o, Wo);

  qk_proj_kernel<<<dim3(32, 6, 2), 256, 0, stream>>>(Xq, Xk, Wq, Wk, Qh, Kh);
  v_proj_kernel<<<dim3(6, 32, 1), 256, 0, stream>>>(Xv, Wv, Vt);
  attn_kernel<<<dim3(768, 1, 1), 128, 0, stream>>>(Qh, Kh, Vt, attn, Ctx);
  o_proj_kernel<<<dim3(32, 6, 1), 256, 0, stream>>>(Ctx, Wo, out);
}
